// Stage2AlignmentModel_65335042507175
// MI455X (gfx1250) — matrix operand order repaired and checked by reading, not compiled, not run on hardware
//
#include <hip/hip_runtime.h>
#include <math.h>

typedef __bf16 bf16_t;
typedef __attribute__((ext_vector_type(16))) __bf16 v16bf;
typedef __attribute__((ext_vector_type(8)))  __bf16 v8bf;
typedef __attribute__((ext_vector_type(8)))  float  v8f;

#define LN_FLAG   1
#define RELU_FLAG 2

// ---------------------------------------------------------------------------
// Generic fused GEMM: C(M x N) = A_bf16(M x Kp) * Wt_bf16(N x Kp)^T + bias
// Block = 256 threads (8 wave32 waves), RB rows per block (16 or 32).
// Waves arranged as (RB/16) row-groups x (8/(RB/16)) col-groups; each wave
// keeps T independent 16x16 accumulators (no WMMA->WMMA RAW chain) and
// register double-buffers the B fragments so the next k-step's global loads
// are in flight while the current step's v_wmma_f32_16x16x32_bf16 execute.
// The contiguous RB x Kp A panel is staged into LDS with
// GLOBAL_LOAD_ASYNC_TO_LDS_B128 (ASYNCcnt) and read as ds_load fragments.
// Epilogue: optional LayerNorm (eps 1e-5), ReLU, per-row scale (mask),
// optional f32 out and/or bf16 out with row stride/offset (seq interleave).
// Host guarantees: M%RB==0, Kp%32==0, N%(colGroups*16*T)==0.
// ---------------------------------------------------------------------------
template <int T, int RB>
__global__ void __launch_bounds__(256)
s2a_gemm(const bf16_t* __restrict__ A, const bf16_t* __restrict__ W,
         const float* __restrict__ bias, const float* __restrict__ rowscale,
         float* __restrict__ outF, bf16_t* __restrict__ outB,
         int Kp, int N, int obStride, int obOffset, int flags)
{
    constexpr int RG  = RB / 16;   // row groups of waves
    constexpr int CG  = 8 / RG;    // col groups of waves
    constexpr int LPR = 256 / RB;  // lanes per row in LN reduction

    extern __shared__ char smem[];
    bf16_t* As   = (bf16_t*)smem;                         // RB * Kp bf16
    float*  Orow = (float*)(smem + (size_t)RB * Kp * 2);  // RB * N f32
    float*  redS = Orow + (size_t)RB * N;                 // 256
    float*  redQ = redS + 256;                            // 256
    float*  mus  = redQ + 256;                            // 32
    float*  rvs  = mus + 32;                              // 32

    const int tid  = threadIdx.x;
    const int w    = tid >> 5;
    const int lane = tid & 31;
    const int m    = lane & 15;   // row (A frag) / col (B frag) within tile
    const int hi   = lane >> 4;
    const int wr   = w % RG;      // wave's row group
    const int wc   = w / RG;      // wave's col group
    const int rowBase = blockIdx.x * RB;

    // ---- async stage of the contiguous RB x Kp A panel into LDS ----
    {
        const char* gsrc = (const char*)(A + (size_t)rowBase * Kp);
        const unsigned ldsBase = (unsigned)(size_t)(void*)As; // low 32b = LDS addr
        const int nChunks = (RB * Kp) / 8;                    // 16-byte chunks
        for (int idx = tid; idx < nChunks; idx += 256) {
            const char* g = gsrc + (size_t)idx * 16;
            unsigned l = ldsBase + (unsigned)idx * 16;
            asm volatile("global_load_async_to_lds_b128 %0, %1, off"
                         :: "v"(l), "v"(g) : "memory");
        }
        asm volatile("s_wait_asynccnt 0x0" ::: "memory");
    }
    __syncthreads();

    const bf16_t* arow = As + (size_t)(16 * wr + m) * Kp;
    const int passCols = CG * 16 * T;
    const int nPass = N / passCols;
    for (int p = 0; p < nPass; ++p) {
        const int waveBase = p * passCols + wc * (16 * T);
        v8f zero = {};
        v8f acc[T];
#pragma unroll
        for (int t = 0; t < T; ++t) acc[t] = zero;
        const bf16_t* brow[T];
#pragma unroll
        for (int t = 0; t < T; ++t)
            brow[t] = W + (size_t)(waveBase + t * 16 + m) * Kp + hi * 16;

        // prime the register double-buffer with k0 = 0 fragments
        v16bf bcur[T];
#pragma unroll
        for (int t = 0; t < T; ++t) bcur[t] = *(const v16bf*)(brow[t]);

        int k0 = 0;
        for (; k0 + 32 < Kp; k0 += 32) {
            // issue next k-step's B loads before consuming the current ones
            v16bf bnxt[T];
#pragma unroll
            for (int t = 0; t < T; ++t)
                bnxt[t] = *(const v16bf*)(brow[t] + k0 + 32);
            // A fragment: lane holds row m; elems 0..15 -> K = k0+hi*16+r
            v8bf alo = *(const v8bf*)(arow + k0 + hi * 16);
            v8bf ahi = *(const v8bf*)(arow + k0 + hi * 16 + 8);
            v16bf a = __builtin_shufflevector(alo, ahi,
                        0,1,2,3,4,5,6,7,8,9,10,11,12,13,14,15);
#pragma unroll
            for (int t = 0; t < T; ++t)
                acc[t] = __builtin_amdgcn_wmma_f32_16x16x32_bf16(
                             false, a, false, bcur[t], (short)0, acc[t], false, false);
#pragma unroll
            for (int t = 0; t < T; ++t) bcur[t] = bnxt[t];
        }
        { // tail k-step
            v8bf alo = *(const v8bf*)(arow + k0 + hi * 16);
            v8bf ahi = *(const v8bf*)(arow + k0 + hi * 16 + 8);
            v16bf a = __builtin_shufflevector(alo, ahi,
                        0,1,2,3,4,5,6,7,8,9,10,11,12,13,14,15);
#pragma unroll
            for (int t = 0; t < T; ++t)
                acc[t] = __builtin_amdgcn_wmma_f32_16x16x32_bf16(
                             false, a, false, bcur[t], (short)0, acc[t], false, false);
        }

#pragma unroll
        for (int t = 0; t < T; ++t) {
            const int col = waveBase + t * 16 + m;
            const float bcol = bias[col];
#pragma unroll
            for (int r = 0; r < 8; ++r)
                Orow[(size_t)(16 * wr + hi * 8 + r) * N + col] = acc[t][r] + bcol;
        }
    }
    __syncthreads();

    if (flags & LN_FLAG) {
        const int r = tid / LPR, l = tid % LPR;
        float s = 0.f, q = 0.f;
        for (int n = l; n < N; n += LPR) {
            float x = Orow[(size_t)r * N + n];
            s += x; q += x * x;
        }
        redS[tid] = s; redQ[tid] = q;
        __syncthreads();
        if (l == 0) {
            float S = 0.f, Q = 0.f;
            for (int j = 0; j < LPR; ++j) { S += redS[r * LPR + j]; Q += redQ[r * LPR + j]; }
            float mean = S / (float)N;
            float var  = Q / (float)N - mean * mean;
            mus[r] = mean;
            rvs[r] = rsqrtf(var + 1e-5f);
        }
        __syncthreads();
    }

    for (int r = 0; r < RB; ++r) {
        const int grow = rowBase + r;
        const float mean = (flags & LN_FLAG) ? mus[r] : 0.f;
        const float rstd = (flags & LN_FLAG) ? rvs[r] : 1.f;
        const float scal = rowscale ? rowscale[grow] : 1.f;
        for (int n = tid; n < N; n += 256) {
            float x = Orow[(size_t)r * N + n];
            if (flags & LN_FLAG)   x = (x - mean) * rstd;
            if (flags & RELU_FLAG) x = fmaxf(x, 0.f);
            x *= scal;
            if (outF) outF[(size_t)grow * N + n] = x;
            if (outB) outB[(size_t)(obOffset + grow * obStride) * N + n] = (bf16_t)x;
        }
    }
}

// ---------------------------------------------------------------------------
// Weight repack: f32 -> bf16 transposed to (N x Kp), zero K-padding.
// srcIsNK==0 : src is (K x N) row-major (reference's  x @ W  layout)
// srcIsNK==1 : src is (N x K) row-major (Wqkv rows / activation convert)
// ---------------------------------------------------------------------------
__global__ void s2a_pack(const float* __restrict__ src, bf16_t* __restrict__ dst,
                         int K, int N, int Kp, int srcIsNK)
{
    long i = (long)blockIdx.x * blockDim.x + threadIdx.x;
    long total = (long)N * Kp;
    if (i >= total) return;
    int n = (int)(i / Kp), k = (int)(i % Kp);
    float v = 0.f;
    if (k < K) v = srcIsNK ? src[(size_t)n * K + k] : src[(size_t)k * N + n];
    dst[i] = (bf16_t)v;
}

// ---------------------------------------------------------------------------
// Build encoder input X (163840 x 64 bf16): [type_emb[type](32), feats(12), 0pad]
// and the per-component validity mask (type > 0).
// ---------------------------------------------------------------------------
__global__ void s2a_build_x(const float* __restrict__ refc, const float* __restrict__ tarc,
                            const float* __restrict__ type_emb,
                            bf16_t* __restrict__ X, float* __restrict__ mask,
                            int total, int half)
{
    int i = blockIdx.x * blockDim.x + threadIdx.x;
    if (i >= total) return;
    const float* c = (i < half) ? (refc + (size_t)i * 13) : (tarc + (size_t)(i - half) * 13);
    int ty = (int)c[0];
    ty = ty < 0 ? 0 : (ty > 19 ? 19 : ty);
    bf16_t* xr = X + (size_t)i * 64;
    const float* te = type_emb + (size_t)ty * 32;
    for (int j = 0; j < 32; ++j) xr[j] = (bf16_t)te[j];
    for (int j = 0; j < 12; ++j) xr[32 + j] = (bf16_t)c[1 + j];
    for (int j = 44; j < 64; ++j) xr[j] = (bf16_t)0.0f;
    mask[i] = (ty > 0) ? 1.0f : 0.0f;
}

// ---------------------------------------------------------------------------
// Per-batch: normalize e_ref/e_tar, sim(20x20) -> d_out, max/argmax, build
// match_features (1025, padded to 1056) directly as bf16 for the Wc GEMM.
// ---------------------------------------------------------------------------
__global__ void __launch_bounds__(128)
s2a_sim(const float* __restrict__ E, float* __restrict__ outSim,
        bf16_t* __restrict__ mfb)
{
    extern __shared__ char smem[];
    float* er   = (float*)smem;        // 20*512
    float* et   = er + 20 * 512;       // 20*512
    float* inr  = et + 20 * 512;       // 20
    float* intr = inr + 20;            // 20
    float* simL = intr + 20;           // 400
    float* sc   = simL + 400;          // 20
    int*   ix   = (int*)(sc + 20);     // 20

    const int b = blockIdx.x, tid = threadIdx.x;
    const float* Er = E + (size_t)b * 20 * 512;
    const float* Et = E + (size_t)(81920 + b * 20) * 512;
    for (int i = tid; i < 20 * 512; i += 128) { er[i] = Er[i]; et[i] = Et[i]; }
    __syncthreads();

    if (tid < 40) {
        int r = tid % 20;
        const float* p = (tid < 20 ? er : et) + (size_t)r * 512;
        float s = 0.f;
        for (int k = 0; k < 512; ++k) s += p[k] * p[k];
        float nn = fmaxf(sqrtf(s), 1e-12f);
        (tid < 20 ? inr : intr)[r] = 1.f / nn;
    }
    __syncthreads();

    for (int i = tid; i < 400; i += 128) {
        int n = i / 20, mm = i % 20;
        const float* a = er + (size_t)n * 512;
        const float* c = et + (size_t)mm * 512;
        float s = 0.f;
        for (int k = 0; k < 512; ++k) s += a[k] * c[k];
        s *= inr[n] * intr[mm];
        simL[i] = s;
        outSim[(size_t)b * 400 + i] = s;
    }
    __syncthreads();

    if (tid < 20) {
        float best = simL[tid * 20]; int bi = 0;
        for (int mm = 1; mm < 20; ++mm) {
            float v = simL[tid * 20 + mm];
            if (v > best) { best = v; bi = mm; } // first-max, matches argmax
        }
        sc[tid] = best; ix[tid] = bi;
    }
    __syncthreads();

    for (int d = tid; d < 1056; d += 128) {
        float v = 0.f;
        if (d < 512) {
            float s = 0.f;
            for (int n = 0; n < 20; ++n) s += er[n * 512 + d];
            v = s * 0.05f;
        } else if (d < 1024) {
            int dd = d - 512; float s = 0.f;
            for (int n = 0; n < 20; ++n)
                if (sc[n] > 0.1f) s += et[ix[n] * 512 + dd];
            v = s * 0.05f;
        } else if (d == 1024) {
            float s = 0.f;
            for (int n = 0; n < 20; ++n) if (sc[n] > 0.1f) s += sc[n];
            v = s * 0.05f;
        }
        mfb[(size_t)b * 1056 + d] = (bf16_t)v;
    }
}

// change_logits = relu(cf @ Wp1 + bp1) @ Wp2 + bp2
__global__ void __launch_bounds__(128)
s2a_logits(const float* __restrict__ cf, const float* __restrict__ Wp1,
           const float* __restrict__ bp1, const float* __restrict__ Wp2,
           const float* __restrict__ bp2, float* __restrict__ outCl)
{
    __shared__ float cfl[1024];
    __shared__ float hl[128];
    const int b = blockIdx.x, tid = threadIdx.x;
    for (int i = tid; i < 1024; i += 128) cfl[i] = cf[(size_t)b * 1024 + i];
    __syncthreads();
    float s = bp1[tid];
    for (int k = 0; k < 1024; ++k) s += cfl[k] * Wp1[(size_t)k * 128 + tid];
    hl[tid] = fmaxf(s, 0.f);
    __syncthreads();
    if (tid < 4) {
        float o = bp2[tid];
        for (int k = 0; k < 128; ++k) o += hl[k] * Wp2[k * 4 + tid];
        outCl[(size_t)b * 4 + tid] = o;
    }
}

// 3-token / 8-head attention: softmax(q.k/sqrt(128)) -> head-mean (3x3).
// ctx / attended are dead code in the reference and skipped.
__global__ void __launch_bounds__(128)
s2a_attn(const float* __restrict__ qk, float* __restrict__ outAw)
{
    __shared__ float q[3 * 1024], kk[3 * 1024], sA[72];
    const int b = blockIdx.x, tid = threadIdx.x;
    for (int i = tid; i < 3 * 1024; i += 128) {
        int row = i >> 10, col = i & 1023;
        const float* base = qk + (size_t)(3 * b + row) * 2048;
        q[i]  = base[col];
        kk[i] = base[1024 + col];
    }
    __syncthreads();
    if (tid < 72) {
        int h = tid / 9, rem = tid % 9, i = rem / 3, j = rem % 3;
        const float* qp = q  + i * 1024 + h * 128;
        const float* kp = kk + j * 1024 + h * 128;
        float s = 0.f;
        for (int d = 0; d < 128; ++d) s += qp[d] * kp[d];
        sA[tid] = s * 0.08838834764831845f; // 1/sqrt(128)
    }
    __syncthreads();
    if (tid < 24) {
        int h = tid / 3, i = tid % 3;
        float* row = sA + h * 9 + i * 3;
        float mx = fmaxf(row[0], fmaxf(row[1], row[2]));
        float e0 = __expf(row[0] - mx), e1 = __expf(row[1] - mx), e2 = __expf(row[2] - mx);
        float inv = 1.f / (e0 + e1 + e2);
        row[0] = e0 * inv; row[1] = e1 * inv; row[2] = e2 * inv;
    }
    __syncthreads();
    if (tid < 9) {
        float s = 0.f;
        for (int h = 0; h < 8; ++h) s += sA[h * 9 + tid];
        outAw[(size_t)b * 9 + tid] = s * 0.125f;
    }
}

// Gating: 3 sigmoid gates over comb(3072), normalize, emit fused_in bf16.
__global__ void __launch_bounds__(256)
s2a_gates(const float* __restrict__ v, const float* __restrict__ t,
          const float* __restrict__ c,
          const float* __restrict__ Wgv, const float* __restrict__ bgv,
          const float* __restrict__ Wgt, const float* __restrict__ bgt,
          const float* __restrict__ Wgc, const float* __restrict__ bgc,
          bf16_t* __restrict__ finb)
{
    __shared__ float comb[3072];
    __shared__ float red[3 * 256];
    __shared__ float g[3];
    const int b = blockIdx.x, tid = threadIdx.x;
    for (int i = tid; i < 1024; i += 256) {
        comb[i]        = v[(size_t)b * 1024 + i];
        comb[1024 + i] = t[(size_t)b * 1024 + i];
        comb[2048 + i] = c[(size_t)b * 1024 + i];
    }
    __syncthreads();
    float p0 = 0.f, p1 = 0.f, p2 = 0.f;
    for (int i = tid; i < 3072; i += 256) {
        float x = comb[i];
        p0 += x * Wgv[i]; p1 += x * Wgt[i]; p2 += x * Wgc[i];
    }
    red[tid] = p0; red[256 + tid] = p1; red[512 + tid] = p2;
    __syncthreads();
    if (tid == 0) {
        float s0 = 0.f, s1 = 0.f, s2 = 0.f;
        for (int j = 0; j < 256; ++j) { s0 += red[j]; s1 += red[256 + j]; s2 += red[512 + j]; }
        float gv = 1.f / (1.f + __expf(-(s0 + bgv[0])));
        float gt = 1.f / (1.f + __expf(-(s1 + bgt[0])));
        float gc = 1.f / (1.f + __expf(-(s2 + bgc[0])));
        float gs = gv + gt + gc + 1e-8f;
        g[0] = gv / gs; g[1] = gt / gs; g[2] = gc / gs;
    }
    __syncthreads();
    for (int i = tid; i < 3072; i += 256)
        finb[(size_t)b * 3072 + i] = (bf16_t)(g[i >> 10] * comb[i]);
}

// ---------------------------------------------------------------------------
static inline size_t s2a_align(size_t x) { return (x + 255) & ~(size_t)255; }

extern "C" void kernel_launch(void* const* d_in, const int* in_sizes, int n_in,
                              void* d_out, int out_size, void* d_ws, size_t ws_size,
                              hipStream_t stream)
{
    (void)in_sizes; (void)n_in; (void)out_size; (void)ws_size;

    const float* refc   = (const float*)d_in[0];
    const float* tarc   = (const float*)d_in[1];
    const float* vis    = (const float*)d_in[2];
    const float* txt    = (const float*)d_in[3];
    const float* temb   = (const float*)d_in[4];
    const float* W1     = (const float*)d_in[5];
    const float* b1     = (const float*)d_in[6];
    const float* W2     = (const float*)d_in[7];
    const float* b2     = (const float*)d_in[8];
    const float* W3     = (const float*)d_in[9];
    const float* b3     = (const float*)d_in[10];
    const float* Wc     = (const float*)d_in[11];
    const float* bc     = (const float*)d_in[12];
    const float* Wp1    = (const float*)d_in[13];
    const float* bp1    = (const float*)d_in[14];
    const float* Wp2    = (const float*)d_in[15];
    const float* bp2    = (const float*)d_in[16];
    const float* Wv     = (const float*)d_in[17];
    const float* bv     = (const float*)d_in[18];
    const float* Wtx    = (const float*)d_in[19];
    const float* bt     = (const float*)d_in[20];
    const float* Wcp    = (const float*)d_in[21];
    const float* bcp    = (const float*)d_in[22];
    const float* Wqkv   = (const float*)d_in[23];
    const float* bqkv   = (const float*)d_in[24];
    // d_in[25]=Wo, d_in[26]=bo : dead code in reference (attended unused)
    const float* Wgv    = (const float*)d_in[27];
    const float* bgv    = (const float*)d_in[28];
    const float* Wgt    = (const float*)d_in[29];
    const float* bgt    = (const float*)d_in[30];
    const float* Wgc    = (const float*)d_in[31];
    const float* bgc    = (const float*)d_in[32];
    const float* Wf1    = (const float*)d_in[33];
    const float* bf1    = (const float*)d_in[34];
    const float* Wf2    = (const float*)d_in[35];
    const float* bf2    = (const float*)d_in[36];

    char* ws = (char*)d_ws;
    size_t off = 0;
    auto alloc = [&](size_t bytes) -> void* {
        void* p = (void*)(ws + off);
        off = s2a_align(off + bytes);
        return p;
    };

    // bf16 transposed/padded weights
    bf16_t* w1t  = (bf16_t*)alloc((size_t)256 * 64 * 2);
    bf16_t* w2t  = (bf16_t*)alloc((size_t)512 * 256 * 2);
    bf16_t* w3t  = (bf16_t*)alloc((size_t)512 * 512 * 2);
    bf16_t* wct  = (bf16_t*)alloc((size_t)1024 * 1056 * 2);
    bf16_t* wvt  = (bf16_t*)alloc((size_t)1024 * 1024 * 2);
    bf16_t* wtt  = (bf16_t*)alloc((size_t)1024 * 1024 * 2);
    bf16_t* wcpt = (bf16_t*)alloc((size_t)1024 * 1024 * 2);
    bf16_t* wqkt = (bf16_t*)alloc((size_t)2048 * 1024 * 2); // Q,K rows only
    bf16_t* wf1t = (bf16_t*)alloc((size_t)2048 * 3072 * 2);
    bf16_t* wf2t = (bf16_t*)alloc((size_t)1024 * 2048 * 2);
    // activations / intermediates
    bf16_t* X    = (bf16_t*)alloc((size_t)163840 * 64 * 2);
    float*  mask = (float*)alloc((size_t)163840 * 4);
    bf16_t* h1b  = (bf16_t*)alloc((size_t)163840 * 256 * 2);
    bf16_t* h2b  = (bf16_t*)alloc((size_t)163840 * 512 * 2);
    float*  E    = (float*)alloc((size_t)163840 * 512 * 4);
    bf16_t* mfb  = (bf16_t*)alloc((size_t)4096 * 1056 * 2);
    float*  cf   = (float*)alloc((size_t)4096 * 1024 * 4);
    bf16_t* cfb  = (bf16_t*)alloc((size_t)4096 * 1024 * 2);
    bf16_t* visb = (bf16_t*)alloc((size_t)4096 * 1024 * 2);
    bf16_t* txtb = (bf16_t*)alloc((size_t)4096 * 1024 * 2);
    float*  vbuf = (float*)alloc((size_t)4096 * 1024 * 4);
    float*  tbuf = (float*)alloc((size_t)4096 * 1024 * 4);
    float*  cbuf = (float*)alloc((size_t)4096 * 1024 * 4);
    bf16_t* seqb = (bf16_t*)alloc((size_t)12288 * 1024 * 2);
    float*  qkb  = (float*)alloc((size_t)12288 * 2048 * 4);
    bf16_t* finb = (bf16_t*)alloc((size_t)4096 * 3072 * 2);
    bf16_t* fb   = (bf16_t*)alloc((size_t)4096 * 2048 * 2);

    float* out      = (float*)d_out;
    float* outFused = out;                                  // 4096 x 1024
    float* outCl    = outFused + (size_t)4096 * 1024;       // 4096 x 4
    float* outSim   = outCl + (size_t)4096 * 4;             // 4096 x 400
    float* outAw    = outSim + (size_t)4096 * 400;          // 4096 x 9

    auto pack = [&](const float* src, bf16_t* dst, int K, int N, int Kp, int srcNK) {
        long total = (long)N * Kp;
        int grid = (int)((total + 255) / 256);
        s2a_pack<<<grid, 256, 0, stream>>>(src, dst, K, N, Kp, srcNK);
    };
    auto gemm = [&](const bf16_t* A, const bf16_t* Wm, const float* bias,
                    const float* rowscale, float* oF, bf16_t* oB,
                    int M, int Kp, int N, int obStride, int obOffset, int flags) {
        const size_t lds32 = (size_t)32 * Kp * 2 + (size_t)32 * N * 4 + 2304;
        const size_t lds16 = (size_t)16 * Kp * 2 + (size_t)16 * N * 4 + 2304;
        if ((M % 32) == 0 && (N % 256) == 0 && lds32 <= (size_t)315 * 1024)
            s2a_gemm<4, 32><<<M / 32, 256, lds32, stream>>>(A, Wm, bias, rowscale, oF, oB,
                                                            Kp, N, obStride, obOffset, flags);
        else if ((N % 512) == 0)
            s2a_gemm<4, 16><<<M / 16, 256, lds16, stream>>>(A, Wm, bias, rowscale, oF, oB,
                                                            Kp, N, obStride, obOffset, flags);
        else
            s2a_gemm<2, 16><<<M / 16, 256, lds16, stream>>>(A, Wm, bias, rowscale, oF, oB,
                                                            Kp, N, obStride, obOffset, flags);
    };

    // ---- weight / activation packing (bf16) ----
    pack(W1,   w1t,  44,   256,  64,   0);
    pack(W2,   w2t,  256,  512,  256,  0);
    pack(W3,   w3t,  512,  512,  512,  0);
    pack(Wc,   wct,  1025, 1024, 1056, 0);
    pack(Wv,   wvt,  1024, 1024, 1024, 0);
    pack(Wtx,  wtt,  1024, 1024, 1024, 0);
    pack(Wcp,  wcpt, 1024, 1024, 1024, 0);
    pack(Wqkv, wqkt, 1024, 2048, 1024, 1); // (3072x1024) rows 0..2047 = Q,K
    pack(Wf1,  wf1t, 3072, 2048, 3072, 0);
    pack(Wf2,  wf2t, 2048, 1024, 2048, 0);
    pack(vis,  visb, 1024, 4096, 1024, 1);
    pack(txt,  txtb, 1024, 4096, 1024, 1);

    // ---- component encoder ----
    s2a_build_x<<<640, 256, 0, stream>>>(refc, tarc, temb, X, mask, 163840, 81920);
    gemm(X,   w1t, b1, nullptr, nullptr, h1b, 163840, 64,  256, 1, 0, LN_FLAG | RELU_FLAG);
    gemm(h1b, w2t, b2, nullptr, nullptr, h2b, 163840, 256, 512, 1, 0, LN_FLAG | RELU_FLAG);
    gemm(h2b, w3t, b3, mask,    E,       nullptr, 163840, 512, 512, 1, 0, 0);

    // ---- matching / match features ----
    {
        size_t lds = (size_t)(20 * 512 * 2 + 20 + 20 + 400 + 20 + 20) * 4;
        s2a_sim<<<4096, 128, lds, stream>>>(E, outSim, mfb);
    }

    // ---- change features + logits ----
    gemm(mfb, wct, bc, nullptr, cf, cfb, 4096, 1056, 1024, 1, 0, LN_FLAG | RELU_FLAG);
    s2a_logits<<<4096, 128, 0, stream>>>(cf, Wp1, bp1, Wp2, bp2, outCl);

    // ---- v / t / c projections (bf16 outputs interleave into seq) ----
    gemm(visb, wvt,  bv,  nullptr, vbuf, seqb, 4096, 1024, 1024, 3, 0, LN_FLAG | RELU_FLAG);
    gemm(txtb, wtt,  bt,  nullptr, tbuf, seqb, 4096, 1024, 1024, 3, 1, LN_FLAG | RELU_FLAG);
    gemm(cfb,  wcpt, bcp, nullptr, cbuf, seqb, 4096, 1024, 1024, 3, 2, LN_FLAG | RELU_FLAG);

    // ---- Q,K projection + attention weights (ctx/attended dead-code removed) ----
    gemm(seqb, wqkt, bqkv, nullptr, qkb, nullptr, 12288, 1024, 2048, 1, 0, 0);
    s2a_attn<<<4096, 128, 0, stream>>>(qkb, outAw);

    // ---- gated fusion ----
    s2a_gates<<<4096, 256, 0, stream>>>(vbuf, tbuf, cbuf, Wgv, bgv, Wgt, bgt, Wgc, bgc, finb);
    gemm(finb, wf1t, bf1, nullptr, nullptr, fb, 4096, 3072, 2048, 1, 0, LN_FLAG | RELU_FLAG);
    gemm(fb,   wf2t, bf2, nullptr, outFused, nullptr, 4096, 2048, 1024, 1, 0, LN_FLAG);
}
